// MultiHeadLatentAttention_78838419685671
// MI455X (gfx1250) — matrix operand order repaired and checked by reading, not compiled, not run on hardware
//
#include <hip/hip_runtime.h>
#include <hip/hip_bf16.h>
#include <math.h>

typedef __bf16 bf16_t;
typedef __attribute__((ext_vector_type(16))) __bf16 v16bf;
typedef __attribute__((ext_vector_type(8)))  float   v8f;

#define B_   2
#define S_   2048
#define E_   1024
#define R_   256
#define H_   16
#define DH_  64
#define DQK_ 128
#define MTOK (B_*S_)   // 4096 tokens

// ---------------------------------------------------------------------------
// CDNA5 async LDS DMA helpers (ASYNCcnt-tracked, per ISA 08_async_tensor)
// LDS operand is the byte offset = low 32 bits of the generic pointer.
// ---------------------------------------------------------------------------
__device__ __forceinline__ void async_ld_b128(void* lds_ptr, const void* gptr) {
  unsigned loff = (unsigned)(size_t)lds_ptr;
  asm volatile("global_load_async_to_lds_b128 %0, %1, off"
               :: "v"(loff), "v"(gptr)
               : "memory");
}
__device__ __forceinline__ void wait_async0() {
  asm volatile("s_wait_asynccnt 0" ::: "memory");
}

// ---------------------------------------------------------------------------
// fp32 -> bf16 conversion
// ---------------------------------------------------------------------------
__global__ void cvt_f32_bf16(const float* __restrict__ src,
                             bf16_t* __restrict__ dst, int n) {
  int i = blockIdx.x * blockDim.x + threadIdx.x;
  if (i < n) dst[i] = (bf16_t)src[i];
}

// ---------------------------------------------------------------------------
// C[M,N] = A[M,K] @ W[N,K]^T   (bf16 in, f32 accumulate, bf16 or f32 out)
// block tile 128x64, 8 waves (4x2), wave tile 32x32 = 2x2 WMMA accumulators.
// K-slab 64, double-buffered LDS filled by global_load_async_to_lds_b128:
//   issue DMA(slab s+1) -> WMMA(slab s) overlapped -> wait_asynccnt -> barrier
// ---------------------------------------------------------------------------
template <bool OUTF32>
__global__ __launch_bounds__(256)
void gemm_bf16(const bf16_t* __restrict__ A, const bf16_t* __restrict__ W,
               void* __restrict__ C, int M, int N, int K) {
  __shared__ __align__(16) bf16_t sA[2][128][72];  // 64 K + 8 pad (16B align)
  __shared__ __align__(16) bf16_t sB[2][64][72];

  const int tid  = threadIdx.x;
  const int lane = tid & 31;
  const int wave = tid >> 5;
  const int wm   = wave & 3;        // wave M tile (0..3)
  const int wn   = wave >> 2;       // wave N tile (0..1)
  const int hf   = lane >> 4;       // lane half (0/1)
  const int l16  = lane & 15;
  const int bm   = blockIdx.x * 128;
  const int bn   = blockIdx.y * 64;

  auto load_slab = [&](int k0, int buf) {
    // A tile 128x64: 1024 x b128, 4 per thread
#pragma unroll
    for (int it = 0; it < 4; ++it) {
      int vid = tid + it * 256;
      int row = vid >> 3;
      int c8  = (vid & 7) << 3;
      async_ld_b128(&sA[buf][row][c8], A + (size_t)(bm + row) * K + k0 + c8);
    }
    // B tile 64x64: 512 x b128, 2 per thread
#pragma unroll
    for (int it = 0; it < 2; ++it) {
      int vid = tid + it * 256;
      int row = vid >> 3;
      int c8  = (vid & 7) << 3;
      async_ld_b128(&sB[buf][row][c8], W + (size_t)(bn + row) * K + k0 + c8);
    }
  };

  v8f acc[2][2];
#pragma unroll
  for (int i = 0; i < 2; ++i)
#pragma unroll
    for (int j = 0; j < 2; ++j)
#pragma unroll
      for (int e = 0; e < 8; ++e) acc[i][j][e] = 0.0f;

  load_slab(0, 0);
  wait_async0();
  __syncthreads();

  const int nslab = K >> 6;
  for (int s = 0; s < nslab; ++s) {
    const int buf = s & 1;
    if (s + 1 < nslab) load_slab((s + 1) << 6, buf ^ 1);  // DMA next slab

#pragma unroll
    for (int kk = 0; kk < 2; ++kk) {
      // A fragments: lane holds M=l16; K = 16*hf..16*hf+15
      v16bf af[2];
#pragma unroll
      for (int i = 0; i < 2; ++i) {
        const bf16_t* p = &sA[buf][wm * 32 + i * 16 + l16][kk * 32];
        ((uint4*)&af[i])[0] = *(const uint4*)(p + 16 * hf);
        ((uint4*)&af[i])[1] = *(const uint4*)(p + 16 * hf + 8);
      }
      // B fragments: lane holds N=l16; K = 16*hf..16*hf+15
      v16bf wf[2];
#pragma unroll
      for (int j = 0; j < 2; ++j) {
        const bf16_t* p = &sB[buf][wn * 32 + j * 16 + l16][kk * 32 + 16 * hf];
        ((uint4*)&wf[j])[0] = ((const uint4*)p)[0];
        ((uint4*)&wf[j])[1] = ((const uint4*)p)[1];
      }
#pragma unroll
      for (int i = 0; i < 2; ++i)
#pragma unroll
        for (int j = 0; j < 2; ++j)
          acc[i][j] = __builtin_amdgcn_wmma_f32_16x16x32_bf16(
              false, af[i], false, wf[j], (short)0, acc[i][j], false, false);
    }
    wait_async0();
    __syncthreads();
  }

  // C layout: vgpr r, lane -> row = r + 8*hf, col = l16
#pragma unroll
  for (int i = 0; i < 2; ++i)
#pragma unroll
    for (int j = 0; j < 2; ++j)
#pragma unroll
      for (int r = 0; r < 8; ++r) {
        int gm = bm + wm * 32 + i * 16 + r + 8 * hf;
        int gn = bn + wn * 32 + j * 16 + l16;
        if (OUTF32)
          ((float*)C)[(size_t)gm * N + gn] = acc[i][j][r];
        else
          ((bf16_t*)C)[(size_t)gm * N + gn] = (bf16_t)acc[i][j][r];
      }
}

// ---------------------------------------------------------------------------
// RoPE + head assembly.
//   q/k out: [B*H, S, 128]  (64 rope dims then 64 nope dims)
//   v   out: [B*H, 64, S]   (transposed so PV B-fragments are contiguous)
// one block (32 threads) per (b,h,s); thread t owns rotation pair (t, t+32)
// ---------------------------------------------------------------------------
__global__ __launch_bounds__(32)
void rope_assemble(const bf16_t* __restrict__ qr_pre, const bf16_t* __restrict__ qn_pre,
                   const bf16_t* __restrict__ kr_pre, const bf16_t* __restrict__ kn_pre,
                   const bf16_t* __restrict__ v_pre,
                   bf16_t* __restrict__ qo, bf16_t* __restrict__ ko,
                   bf16_t* __restrict__ vo) {
  const int bh = blockIdx.x >> 11;   // 0..31
  const int s  = blockIdx.x & 2047;
  const int b  = bh >> 4;
  const int h  = bh & 15;
  const int t  = threadIdx.x;        // 0..31

  const size_t tok = ((size_t)b * S_ + s) * E_ + h * 64;

  // inv_freq = 10000^(-2t/64); ang = s * inv_freq
  const float ang = (float)s * __expf(-(2.0f * (float)t / 64.0f) * 9.210340371976184f);
  float sn, cs;
  __sincosf(ang, &sn, &cs);

  const float q1 = (float)qr_pre[tok + t], q2 = (float)qr_pre[tok + t + 32];
  const float k1 = (float)kr_pre[tok + t], k2 = (float)kr_pre[tok + t + 32];

  const size_t qk = ((size_t)bh * S_ + s) * DQK_;
  qo[qk + t]           = (bf16_t)(q1 * cs - q2 * sn);
  qo[qk + t + 32]      = (bf16_t)(q2 * cs + q1 * sn);
  qo[qk + 64 + t]      = qn_pre[tok + t];
  qo[qk + 64 + t + 32] = qn_pre[tok + t + 32];

  ko[qk + t]           = (bf16_t)(k1 * cs - k2 * sn);
  ko[qk + t + 32]      = (bf16_t)(k2 * cs + k1 * sn);
  ko[qk + 64 + t]      = kn_pre[tok + t];
  ko[qk + 64 + t + 32] = kn_pre[tok + t + 32];

  vo[((size_t)bh * 64 + t) * S_ + s]      = v_pre[tok + t];
  vo[((size_t)bh * 64 + t + 32) * S_ + s] = v_pre[tok + t + 32];
}

// ---------------------------------------------------------------------------
// Causal flash attention, online softmax.
// grid = (B*H, S/64), block = 128 threads = 4 waves; each wave owns 16 q rows.
// K/V tiles double-buffered in LDS via global_load_async_to_lds_b128.
// Scores and PV both via v_wmma_f32_16x16x32_bf16.
// ---------------------------------------------------------------------------
__global__ __launch_bounds__(128)
void flash_attn(const bf16_t* __restrict__ Q, const bf16_t* __restrict__ Kc,
                const bf16_t* __restrict__ Vt, bf16_t* __restrict__ Y) {
  __shared__ __align__(16) bf16_t sK[2][32][136];  // 32 keys x 128 dims (+pad)
  __shared__ __align__(16) bf16_t sV[2][64][40];   // 64 dims x 32 keys (+pad)
  __shared__ __align__(16) bf16_t sP[4][16][40];   // per-wave P staging 16x32

  const int bh   = blockIdx.x;
  const int b    = bh >> 4;
  const int h    = bh & 15;
  const int qb   = blockIdx.y * 64;
  const int tid  = threadIdx.x;
  const int lane = tid & 31;
  const int wave = tid >> 5;
  const int hf   = lane >> 4;
  const int l16  = lane & 15;
  const float scale = 0.08838834764831845f;  // 1/sqrt(128)

  const bf16_t* kbase = Kc + (size_t)bh * S_ * DQK_;
  const bf16_t* vbase = Vt + (size_t)bh * 64 * S_;

  auto load_kv = [&](int jt, int buf) {
    // K tile 32x128: 512 x b128, 4 per thread
#pragma unroll
    for (int it = 0; it < 4; ++it) {
      int vid = tid + it * 128;
      int row = vid >> 4;
      int c8  = (vid & 15) << 3;
      async_ld_b128(&sK[buf][row][c8],
                    kbase + (size_t)(jt * 32 + row) * DQK_ + c8);
    }
    // V tile (transposed) 64x32: 256 x b128, 2 per thread
#pragma unroll
    for (int it = 0; it < 2; ++it) {
      int vid = tid + it * 128;
      int row = vid >> 2;
      int c8  = (vid & 3) << 3;
      async_ld_b128(&sV[buf][row][c8],
                    vbase + (size_t)row * S_ + jt * 32 + c8);
    }
  };

  // Q A-fragments, 4 chunks over d=128 (loaded once, from global)
  v16bf aq[4];
  {
    const bf16_t* qp = Q + ((size_t)bh * S_ + (qb + wave * 16 + l16)) * DQK_;
#pragma unroll
    for (int cch = 0; cch < 4; ++cch) {
      ((uint4*)&aq[cch])[0] = *(const uint4*)(qp + cch * 32 + 16 * hf);
      ((uint4*)&aq[cch])[1] = *(const uint4*)(qp + cch * 32 + 16 * hf + 8);
    }
  }

  float mrow[8], lrow[8];
  v8f accy[4];
#pragma unroll
  for (int r = 0; r < 8; ++r) { mrow[r] = -1e30f; lrow[r] = 0.0f; }
#pragma unroll
  for (int t = 0; t < 4; ++t)
#pragma unroll
    for (int e = 0; e < 8; ++e) accy[t][e] = 0.0f;

  const int jt_max = (qb + 64) / 32;   // causal: key tiles up to q block end

  load_kv(0, 0);
  wait_async0();
  __syncthreads();

  for (int jt = 0; jt < jt_max; ++jt) {
    const int buf = jt & 1;
    if (jt + 1 < jt_max) load_kv(jt + 1, buf ^ 1);  // DMA next K/V tile

    // scores = Q @ K^T for 16 q rows x 32 keys
    v8f c0, c1;
#pragma unroll
    for (int e = 0; e < 8; ++e) { c0[e] = 0.0f; c1[e] = 0.0f; }
#pragma unroll
    for (int cch = 0; cch < 4; ++cch) {
      v16bf b0, b1;
      const bf16_t* p0 = &sK[buf][l16][cch * 32 + 16 * hf];
      ((uint4*)&b0)[0] = ((const uint4*)p0)[0];
      ((uint4*)&b0)[1] = ((const uint4*)p0)[1];
      const bf16_t* p1 = &sK[buf][16 + l16][cch * 32 + 16 * hf];
      ((uint4*)&b1)[0] = ((const uint4*)p1)[0];
      ((uint4*)&b1)[1] = ((const uint4*)p1)[1];
      c0 = __builtin_amdgcn_wmma_f32_16x16x32_bf16(false, aq[cch], false, b0,
                                                   (short)0, c0, false, false);
      c1 = __builtin_amdgcn_wmma_f32_16x16x32_bf16(false, aq[cch], false, b1,
                                                   (short)0, c1, false, false);
    }

    // online softmax per row (row = r + 8*hf; columns live across 16 lanes)
#pragma unroll
    for (int r = 0; r < 8; ++r) {
      int qg  = qb + wave * 16 + r + 8 * hf;
      int k0i = jt * 32 + l16;
      int k1i = k0i + 16;
      float s0 = c0[r] * scale; if (k0i > qg) s0 = -1e30f;
      float s1 = c1[r] * scale; if (k1i > qg) s1 = -1e30f;
      float mx = fmaxf(s0, s1);
      mx = fmaxf(mx, __shfl_xor(mx, 1));
      mx = fmaxf(mx, __shfl_xor(mx, 2));
      mx = fmaxf(mx, __shfl_xor(mx, 4));
      mx = fmaxf(mx, __shfl_xor(mx, 8));
      float mnew = fmaxf(mrow[r], mx);
      float p0 = __expf(s0 - mnew);
      float p1 = __expf(s1 - mnew);
      float rs = p0 + p1;
      rs += __shfl_xor(rs, 1);
      rs += __shfl_xor(rs, 2);
      rs += __shfl_xor(rs, 4);
      rs += __shfl_xor(rs, 8);
      float alpha = __expf(mrow[r] - mnew);
      mrow[r] = mnew;
      lrow[r] = lrow[r] * alpha + rs;
#pragma unroll
      for (int t = 0; t < 4; ++t) accy[t][r] *= alpha;
      // redistribute P (C layout) into LDS for A-fragment reload
      sP[wave][r + 8 * hf][l16]      = (bf16_t)p0;
      sP[wave][r + 8 * hf][16 + l16] = (bf16_t)p1;
    }
    __syncthreads();

    // P A-fragment (16x32) and V B-fragments (32 keys x 16 dims per tile)
    v16bf ap;
    {
      const bf16_t* pp = &sP[wave][l16][0];
      ((uint4*)&ap)[0] = *(const uint4*)(pp + 16 * hf);
      ((uint4*)&ap)[1] = *(const uint4*)(pp + 16 * hf + 8);
    }
#pragma unroll
    for (int t = 0; t < 4; ++t) {
      v16bf bv;
      const bf16_t* vp = &sV[buf][t * 16 + l16][16 * hf];
      ((uint4*)&bv)[0] = ((const uint4*)vp)[0];
      ((uint4*)&bv)[1] = ((const uint4*)vp)[1];
      accy[t] = __builtin_amdgcn_wmma_f32_16x16x32_bf16(false, ap, false, bv,
                                                        (short)0, accy[t],
                                                        false, false);
    }

    wait_async0();       // next tile's DMA complete
    __syncthreads();     // all waves done with this tile + sP
  }

  // normalize and write y: [B, S, E] bf16
#pragma unroll
  for (int r = 0; r < 8; ++r) {
    float inv = 1.0f / lrow[r];
    int qg = qb + wave * 16 + r + 8 * hf;
    size_t base = ((size_t)b * S_ + qg) * E_ + h * 64;
#pragma unroll
    for (int t = 0; t < 4; ++t)
      Y[base + t * 16 + l16] = (bf16_t)(accy[t][r] * inv);
  }
}

// ---------------------------------------------------------------------------
// host-side orchestration
// ---------------------------------------------------------------------------
extern "C" void kernel_launch(void* const* d_in, const int* in_sizes, int n_in,
                              void* d_out, int out_size, void* d_ws, size_t ws_size,
                              hipStream_t stream) {
  (void)in_sizes; (void)n_in; (void)out_size; (void)ws_size;

  const float* x        = (const float*)d_in[0];
  const float* wq_down  = (const float*)d_in[1];
  const float* wk_rope  = (const float*)d_in[2];
  const float* wkv_down = (const float*)d_in[3];
  const float* wq_rope  = (const float*)d_in[4];
  const float* wq_up    = (const float*)d_in[5];
  const float* wk_up    = (const float*)d_in[6];
  const float* wv_up    = (const float*)d_in[7];
  const float* wo       = (const float*)d_in[8];

  char*  ws  = (char*)d_ws;
  size_t off = 0;
  auto alloc = [&](size_t bytes) -> void* {
    void* p = ws + off;
    off = (off + bytes + 255) & ~(size_t)255;
    return p;
  };

  bf16_t* xb     = (bf16_t*)alloc((size_t)MTOK * E_ * 2);
  bf16_t* wqd_b  = (bf16_t*)alloc((size_t)R_ * E_ * 2);
  bf16_t* wkr_b  = (bf16_t*)alloc((size_t)R_ * E_ * 2);
  bf16_t* wkvd_b = (bf16_t*)alloc((size_t)R_ * E_ * 2);
  bf16_t* wqr_b  = (bf16_t*)alloc((size_t)E_ * R_ * 2);
  bf16_t* wqu_b  = (bf16_t*)alloc((size_t)E_ * R_ * 2);
  bf16_t* wku_b  = (bf16_t*)alloc((size_t)E_ * R_ * 2);
  bf16_t* wvu_b  = (bf16_t*)alloc((size_t)E_ * R_ * 2);
  bf16_t* wo_b   = (bf16_t*)alloc((size_t)E_ * E_ * 2);
  bf16_t* qlat   = (bf16_t*)alloc((size_t)MTOK * R_ * 2);
  bf16_t* ckr    = (bf16_t*)alloc((size_t)MTOK * R_ * 2);
  bf16_t* kvlat  = (bf16_t*)alloc((size_t)MTOK * R_ * 2);
  bf16_t* qr_pre = (bf16_t*)alloc((size_t)MTOK * E_ * 2);
  bf16_t* qn_pre = (bf16_t*)alloc((size_t)MTOK * E_ * 2);
  bf16_t* kr_pre = (bf16_t*)alloc((size_t)MTOK * E_ * 2);
  bf16_t* kn_pre = (bf16_t*)alloc((size_t)MTOK * E_ * 2);
  bf16_t* v_pre  = (bf16_t*)alloc((size_t)MTOK * E_ * 2);
  bf16_t* qasm   = (bf16_t*)alloc((size_t)B_ * H_ * S_ * DQK_ * 2);
  bf16_t* kasm   = (bf16_t*)alloc((size_t)B_ * H_ * S_ * DQK_ * 2);
  bf16_t* vasm   = (bf16_t*)alloc((size_t)B_ * H_ * DH_ * S_ * 2);
  bf16_t* ybuf   = (bf16_t*)alloc((size_t)MTOK * E_ * 2);

  auto cvt = [&](const float* s, bf16_t* d, size_t n) {
    cvt_f32_bf16<<<dim3((unsigned)((n + 255) / 256)), dim3(256), 0, stream>>>(
        s, d, (int)n);
  };
  cvt(x,        xb,     (size_t)MTOK * E_);
  cvt(wq_down,  wqd_b,  (size_t)R_ * E_);
  cvt(wk_rope,  wkr_b,  (size_t)R_ * E_);
  cvt(wkv_down, wkvd_b, (size_t)R_ * E_);
  cvt(wq_rope,  wqr_b,  (size_t)E_ * R_);
  cvt(wq_up,    wqu_b,  (size_t)E_ * R_);
  cvt(wk_up,    wku_b,  (size_t)E_ * R_);
  cvt(wv_up,    wvu_b,  (size_t)E_ * R_);
  cvt(wo,       wo_b,   (size_t)E_ * E_);

  auto gemm = [&](const bf16_t* A, const bf16_t* W, void* C, int M, int N,
                  int K, bool f32out) {
    dim3 grid(M / 128, N / 64);
    if (f32out)
      gemm_bf16<true><<<grid, dim3(256), 0, stream>>>(A, W, C, M, N, K);
    else
      gemm_bf16<false><<<grid, dim3(256), 0, stream>>>(A, W, C, M, N, K);
  };

  // latent projections (K = E = 1024)
  gemm(xb, wqd_b,  qlat,  MTOK, R_, E_, false);
  gemm(xb, wkr_b,  ckr,   MTOK, R_, E_, false);
  gemm(xb, wkvd_b, kvlat, MTOK, R_, E_, false);
  // up projections (K = R = 256)
  gemm(qlat,  wqr_b, qr_pre, MTOK, E_, R_, false);
  gemm(qlat,  wqu_b, qn_pre, MTOK, E_, R_, false);
  gemm(ckr,   wku_b, kr_pre, MTOK, E_, R_, false);
  gemm(kvlat, wku_b, kn_pre, MTOK, E_, R_, false);
  gemm(kvlat, wvu_b, v_pre,  MTOK, E_, R_, false);

  rope_assemble<<<dim3(B_ * H_ * S_), dim3(32), 0, stream>>>(
      qr_pre, qn_pre, kr_pre, kn_pre, v_pre, qasm, kasm, vasm);

  flash_attn<<<dim3(B_ * H_, S_ / 64), dim3(128), 0, stream>>>(qasm, kasm,
                                                               vasm, ybuf);

  // final output projection: f32 result into d_out
  gemm(ybuf, wo_b, d_out, MTOK, E_, E_, true);
}
